// NasCoraCell_36816459661697
// MI455X (gfx1250) — matrix operand order repaired and checked by reading, not compiled, not run on hardware
//
#include <hip/hip_runtime.h>
#include <hip/hip_bf16.h>
#include <math.h>

// ---------------- problem constants (match reference) ----------------
#define NN   20000      // nodes
#define EE   320000     // edges
#define CURD 256
#define HIDD 256
#define OUTD 64
#define HEADS 6
#define GATF (HEADS*OUTD)    // 384
#define CATF (GATF + 3*OUTD) // 576

typedef __attribute__((ext_vector_type(16))) _Float16 v16h;
typedef __attribute__((ext_vector_type(8)))  float    v8f;

// ---------------- helpers ----------------
__device__ __forceinline__ unsigned f2o(float f) {
    unsigned u = __float_as_uint(f);
    return (u & 0x80000000u) ? ~u : (u | 0x80000000u);
}
__device__ __forceinline__ float o2f(unsigned u) {
    return __uint_as_float((u & 0x80000000u) ? (u & 0x7FFFFFFFu) : ~u);
}
__device__ __forceinline__ float lrelu(float v, float s) { return v > 0.0f ? v : s * v; }

// ---------------- fill ----------------
__global__ void fill_f32(float* __restrict__ p, float v, int n) {
    int i = blockIdx.x * blockDim.x + threadIdx.x;
    if (i < n) p[i] = v;
}

// ---------------- WMMA GEMM ----------------
// C[M,Nc] = A[M,K] @ B[K,Nc] (+bias per col), f32 in/out, f16 WMMA fragments.
//
// Block = 256 threads = 8 waves. Macro-tile: 128 rows x 64 cols.
//   wave w -> rows m0 = blockIdx.x*128 + w*16
//   cols   -> n0 = blockIdx.y*64, four 16-col WMMA tiles per wave.
// Per k-step (K += 32):
//   - all 256 threads stage the 32x64 B slab into LDS, converting f32->f16 and
//     pre-swizzling into B-fragment order: lds[((c*32 + n + 16*g))*16 + ((k&7) | (k>>4)<<3)], g = (k>>3)&1
//     so a fragment is 16 contiguous halves per lane (two ds_load_b128).
//   - each wave builds its A fragment from global (row-contiguous, vectorizes
//     to global_load_b128), prefetches the next A k-block, preloads ALL four
//     B fragments (8x ds_load_b128 in flight), then issues 4 back-to-back
//     WMMAs so the matrix pipe overlaps remaining DS latency.
//
// A fragment layout (ISA 7.12.2, 16-bit A 16x32): lane = m + 16*g holds
//   halves j: k = (q<4 ? 2q : 16+2(q-4)) + 8g + (j&1), q=j>>1
// B fragment (16-bit B 32x16): lane = n + 16*g holds k = (j<8 ? j : 8+j) + 8*g in half j.
// C/D: VGPR i -> row m0 + i + 8*(lane>=16), col = n0 + (lane&15).
__global__ void __launch_bounds__(256)
wmma_gemm(const float* __restrict__ A, const float* __restrict__ B,
          const float* __restrict__ bias, float* __restrict__ C,
          int M, int Nc, int K) {
    __shared__ _Float16 ldsB[4 * 32 * 16];   // 4 ctiles x 32 lanes x 16 halves = 4 KB

    const int tid  = threadIdx.x;
    const int lane = tid & 31;
    const int wave = tid >> 5;
    const int m0   = blockIdx.x * 128 + wave * 16;
    const int n0   = blockIdx.y * 64;
    const int r    = lane & 15;              // A row in tile / B col in tile
    const int g    = lane >> 4;
    const bool active = (m0 < M);            // whole-wave predicate (M % 16 == 0)

    // cooperative B-fill mapping: thread -> (k row, 8 consecutive cols)
    const int fk   = tid >> 3;               // 0..31
    const int fc0  = (tid & 7) * 8;          // 0,8,...,56
    const int fg   = (fk >> 3) & 1, fkk = (fk & 7) | ((fk >> 4) << 3);

    v8f acc[4] = {};
    for (int k0 = 0; k0 < K; k0 += 32) {
        __syncthreads();
        {   // stage + swizzle B slab (coalesced 32B row reads, b16 LDS scatter)
            const float* brow = B + (size_t)(k0 + fk) * Nc + n0 + fc0;
            #pragma unroll
            for (int u = 0; u < 8; ++u) {
                const int col = fc0 + u;
                ldsB[((((col >> 4) * 32) + (col & 15) + 16 * fg) << 4) + fkk] =
                    (_Float16)brow[u];
            }
        }
        __syncthreads();
        if (!active) continue;

        v16h a;
        const float* arow = A + (size_t)(m0 + r) * K + k0 + 8 * g;
        #pragma unroll
        for (int p = 0; p < 8; ++p) {
            const int kk = (p < 4) ? (2 * p) : (16 + 2 * (p - 4));
            const float2 f2 = *(const float2*)(arow + kk);
            a[2 * p]     = (_Float16)f2.x;
            a[2 * p + 1] = (_Float16)f2.y;
        }
        __builtin_prefetch(arow + 32, 0, 1);  // next A k-block -> global_prefetch

        // preload all four B fragments, then issue WMMAs back-to-back
        v16h bfr[4];
        #pragma unroll
        for (int c = 0; c < 4; ++c) {
            const _Float16* lp = &ldsB[(c * 32 + lane) << 4];  // 32B aligned
            #pragma unroll
            for (int j = 0; j < 16; ++j) bfr[c][j] = lp[j];
        }
        #pragma unroll
        for (int c = 0; c < 4; ++c)
            acc[c] = __builtin_amdgcn_wmma_f32_16x16x32_f16(
                false, a, false, bfr[c], (short)0, acc[c], false, false);
    }

    if (!active) return;
    const int mb = m0 + (g ? 8 : 0);
    #pragma unroll
    for (int c = 0; c < 4; ++c) {
        const int col = n0 + c * 16 + r;
        const float bv = bias ? bias[col] : 0.0f;
        #pragma unroll
        for (int i = 0; i < 8; ++i)
            C[(size_t)(mb + i) * Nc + col] = acc[c][i] + bv;
    }
}

// ---------------- GAT ----------------
// Per (node, head): al = <xw, att_src>, ar = <xw, att_dst>, init max with self-loop e.
__global__ void gat_node_kernel(const float* __restrict__ xwg,
                                const float* __restrict__ att_s,
                                const float* __restrict__ att_d,
                                float* __restrict__ al, float* __restrict__ ar,
                                unsigned* __restrict__ m) {
    int idx = blockIdx.x * blockDim.x + threadIdx.x;   // n*HEADS + h
    if (idx >= NN * HEADS) return;
    const int n = idx / HEADS, h = idx % HEADS;
    const float* xr = xwg + (size_t)n * GATF + h * OUTD;
    const float* as = att_s + h * OUTD;
    const float* ad = att_d + h * OUTD;
    float sa = 0.0f, sd = 0.0f;
    for (int f = 0; f < OUTD; ++f) { const float v = xr[f]; sa += v * as[f]; sd += v * ad[f]; }
    al[idx] = sa; ar[idx] = sd;
    m[idx] = f2o(lrelu(sa + sd, 0.2f));   // self-loop edge seeds the segment max
}

__global__ void gat_max_kernel(const int* __restrict__ src, const int* __restrict__ dst,
                               const float* __restrict__ al, const float* __restrict__ ar,
                               unsigned* __restrict__ m) {
    int e = blockIdx.x * blockDim.x + threadIdx.x;
    if (e >= EE) return;
    const int s = src[e], d = dst[e];
    #pragma unroll
    for (int h = 0; h < HEADS; ++h) {
        const float v = lrelu(al[s * HEADS + h] + ar[d * HEADS + h], 0.2f);
        atomicMax(&m[d * HEADS + h], f2o(v));
    }
}

// Wave per extended edge (e >= EE are self loops). Accumulate denom and pe*xw[s].
__global__ void __launch_bounds__(256)
gat_accum_kernel(const int* __restrict__ src, const int* __restrict__ dst,
                 const float* __restrict__ al, const float* __restrict__ ar,
                 const unsigned* __restrict__ m, const float* __restrict__ xwg,
                 float* __restrict__ den, float* __restrict__ num) {
    const int lane = threadIdx.x & 31;
    const int e = blockIdx.x * 8 + (threadIdx.x >> 5);
    if (e >= EE + NN) return;
    int s, d;
    if (e < EE) { s = src[e]; d = dst[e]; } else { s = d = e - EE; }
    float pe[HEADS];
    #pragma unroll
    for (int h = 0; h < HEADS; ++h) {
        const float v = lrelu(al[s * HEADS + h] + ar[d * HEADS + h], 0.2f);
        pe[h] = __expf(v - o2f(m[d * HEADS + h]));
    }
    #pragma unroll
    for (int h = 0; h < HEADS; ++h)
        if (lane == h) atomicAdd(&den[d * HEADS + h], pe[h]);
    const float* xr = xwg + (size_t)s * GATF;
    float* nr = num + (size_t)d * GATF;
    #pragma unroll
    for (int it = 0; it < GATF / 32; ++it) {        // head = it>>1 (64 = 2*32)
        const int f = lane + 32 * it;
        atomicAdd(&nr[f], pe[it >> 1] * xr[f]);
    }
}

// num -> h1 = lrelu(num/den + b_gat, 0.01), in place.
__global__ void gat_final_kernel(float* __restrict__ num, const float* __restrict__ den,
                                 const float* __restrict__ b_gat) {
    int idx = blockIdx.x * blockDim.x + threadIdx.x;
    if (idx >= NN * GATF) return;
    const int n = idx / GATF, c = idx % GATF;
    const float v = num[idx] / den[n * HEADS + (c >> 6)] + b_gat[c];
    num[idx] = lrelu(v, 0.01f);
}

// ---------------- degrees ----------------
__global__ void deg_kernel(const int* __restrict__ dst, const float* __restrict__ w,
                           float* __restrict__ deg, float* __restrict__ dega) {
    int e = blockIdx.x * blockDim.x + threadIdx.x;
    if (e >= EE) return;
    const int d = dst[e];
    atomicAdd(&deg[d], w[e]);    // deg pre-initialized to 1.0 (self loop weight)
    atomicAdd(&dega[d], w[e]);   // ARMA: no self loops, pre-initialized to 0
}

// ---------------- GCN scatter (both convs, extended edges) ----------------
__global__ void __launch_bounds__(256)
gcn_accum_kernel(const int* __restrict__ src, const int* __restrict__ dst,
                 const float* __restrict__ w, const float* __restrict__ deg,
                 const float* __restrict__ xw0, const float* __restrict__ xw1,
                 float* __restrict__ agg0, float* __restrict__ agg1) {
    const int lane = threadIdx.x & 31;
    const int e = blockIdx.x * 8 + (threadIdx.x >> 5);
    if (e >= EE + NN) return;
    int s, d; float wt;
    if (e < EE) { s = src[e]; d = dst[e]; wt = w[e]; } else { s = d = e - EE; wt = 1.0f; }
    const float ds = deg[s], dd = deg[d];
    const float norm = (ds > 0.0f ? rsqrtf(ds) : 0.0f) * wt * (dd > 0.0f ? rsqrtf(dd) : 0.0f);
    #pragma unroll
    for (int it = 0; it < OUTD / 32; ++it) {
        const int f = lane + 32 * it;
        atomicAdd(&agg0[(size_t)d * OUTD + f], norm * xw0[(size_t)s * OUTD + f]);
        atomicAdd(&agg1[(size_t)d * OUTD + f], norm * xw1[(size_t)s * OUTD + f]);
    }
}

// ---------------- ARMA scatter (original edges only) ----------------
__global__ void __launch_bounds__(256)
arma_accum_kernel(const int* __restrict__ src, const int* __restrict__ dst,
                  const float* __restrict__ w, const float* __restrict__ dega,
                  const float* __restrict__ t, float* __restrict__ agga) {
    const int lane = threadIdx.x & 31;
    const int e = blockIdx.x * 8 + (threadIdx.x >> 5);
    if (e >= EE) return;
    const int s = src[e], d = dst[e];
    const float ds = dega[s], dd = dega[d];
    const float norm = (ds > 0.0f ? rsqrtf(ds) : 0.0f) * w[e] * (dd > 0.0f ? rsqrtf(dd) : 0.0f);
    #pragma unroll
    for (int it = 0; it < OUTD / 32; ++it) {
        const int f = lane + 32 * it;
        atomicAdd(&agga[(size_t)d * OUTD + f], norm * t[(size_t)s * OUTD + f]);
    }
}

// ---------------- final concat + activations + tanh ----------------
__global__ void final_kernel(const float* __restrict__ h1, const float* __restrict__ agg0,
                             const float* __restrict__ agg1, const float* __restrict__ agga,
                             const float* __restrict__ root,
                             const float* __restrict__ b0, const float* __restrict__ b1,
                             const float* __restrict__ ba, float* __restrict__ out) {
    int idx = blockIdx.x * blockDim.x + threadIdx.x;
    if (idx >= NN * CATF) return;
    const int n = idx / CATF, j = idx % CATF;
    float v;
    if (j < GATF) {
        v = h1[(size_t)n * GATF + j];                                   // already lrelu'd
    } else if (j < GATF + OUTD) {
        const int c = j - GATF;
        v = lrelu(agg0[(size_t)n * OUTD + c] + b0[c], 0.01f);
    } else if (j < GATF + 2 * OUTD) {
        const int c = j - GATF - OUTD;
        v = lrelu(agg1[(size_t)n * OUTD + c] + b1[c], 0.01f);
    } else {
        const int c = j - GATF - 2 * OUTD;
        v = agga[(size_t)n * OUTD + c] + root[(size_t)n * OUTD + c] + ba[c];
        v = v > 0.0f ? v : 0.0f;                                        // relu (lrelu is no-op)
    }
    out[idx] = tanhf(v);
}

// ---------------- host launch ----------------
static inline int cdiv(long long a, long long b) { return (int)((a + b - 1) / b); }

extern "C" void kernel_launch(void* const* d_in, const int* in_sizes, int n_in,
                              void* d_out, int out_size, void* d_ws, size_t ws_size,
                              hipStream_t stream) {
    const float* x      = (const float*)d_in[0];
    const int*   ei     = (const int*)  d_in[1];
    const float* ew     = (const float*)d_in[2];
    const float* Wp     = (const float*)d_in[3];
    const float* bp     = (const float*)d_in[4];
    const float* Wgat   = (const float*)d_in[5];
    const float* att_s  = (const float*)d_in[6];
    const float* att_d  = (const float*)d_in[7];
    const float* b_gat  = (const float*)d_in[8];
    const float* Wg0    = (const float*)d_in[9];
    const float* bg0    = (const float*)d_in[10];
    const float* Wg1    = (const float*)d_in[11];
    const float* bg1    = (const float*)d_in[12];
    const float* Wai    = (const float*)d_in[13];
    const float* War    = (const float*)d_in[14];
    const float* ba     = (const float*)d_in[15];
    const int* src = ei;
    const int* dst = ei + EE;
    float* out = (float*)d_out;

    // workspace carve-up
    float* ws = (float*)d_ws;
    size_t off = 0;
    auto grab = [&](size_t n) { float* p = ws + off; off += n; return p; };
    float*    h    = grab((size_t)NN * HIDD);
    float*    xwg  = grab((size_t)NN * GATF);
    float*    num  = grab((size_t)NN * GATF);   // becomes h1 in place
    float*    al   = grab((size_t)NN * HEADS);
    float*    ar   = grab((size_t)NN * HEADS);
    unsigned* m    = (unsigned*)grab((size_t)NN * HEADS);
    float*    den  = grab((size_t)NN * HEADS);
    float*    xw0  = grab((size_t)NN * OUTD);
    float*    xw1  = grab((size_t)NN * OUTD);
    float*    agg0 = grab((size_t)NN * OUTD);
    float*    agg1 = grab((size_t)NN * OUTD);
    float*    t    = grab((size_t)NN * OUTD);
    float*    root = grab((size_t)NN * OUTD);
    float*    agga = grab((size_t)NN * OUTD);
    float*    deg  = grab(NN);
    float*    dega = grab(NN);
    (void)ws_size; (void)n_in; (void)in_sizes; (void)out_size;

    const int TB = 256;
    // 1) init accumulators
    fill_f32<<<cdiv((size_t)NN * GATF, TB), TB, 0, stream>>>(num, 0.0f, NN * GATF);
    fill_f32<<<cdiv(NN * HEADS, TB), TB, 0, stream>>>(den, 0.0f, NN * HEADS);
    fill_f32<<<cdiv(NN * OUTD, TB), TB, 0, stream>>>(agg0, 0.0f, NN * OUTD);
    fill_f32<<<cdiv(NN * OUTD, TB), TB, 0, stream>>>(agg1, 0.0f, NN * OUTD);
    fill_f32<<<cdiv(NN * OUTD, TB), TB, 0, stream>>>(agga, 0.0f, NN * OUTD);
    fill_f32<<<cdiv(NN, TB), TB, 0, stream>>>(deg, 1.0f, NN);   // self-loop weight 1
    fill_f32<<<cdiv(NN, TB), TB, 0, stream>>>(dega, 0.0f, NN);

    // 2) projection: h = x @ Wp + bp
    wmma_gemm<<<dim3(cdiv(NN, 128), HIDD / 64), TB, 0, stream>>>(x, Wp, bp, h, NN, HIDD, CURD);
    // 3) GAT transform: xwg = h @ Wgat (bias added post-aggregation)
    wmma_gemm<<<dim3(cdiv(NN, 128), GATF / 64), TB, 0, stream>>>(h, Wgat, nullptr, xwg, NN, GATF, HIDD);
    // 4) per-node attention terms + self-loop max seed
    gat_node_kernel<<<cdiv(NN * HEADS, TB), TB, 0, stream>>>(xwg, att_s, att_d, al, ar, m);
    // 5) segment max over incoming edges
    gat_max_kernel<<<cdiv(EE, TB), TB, 0, stream>>>(src, dst, al, ar, m);
    // 6) softmax numerator/denominator scatter (wave per extended edge)
    gat_accum_kernel<<<cdiv(EE + NN, 8), TB, 0, stream>>>(src, dst, al, ar, m, xwg, den, num);
    // 7) h1 = lrelu(num/den + b_gat) in place
    gat_final_kernel<<<cdiv((size_t)NN * GATF, TB), TB, 0, stream>>>(num, den, b_gat);
    float* h1 = num;

    // 8) remaining GEMMs (Nc = 64 -> gridDim.y = 1)
    wmma_gemm<<<dim3(cdiv(NN, 128), OUTD / 64), TB, 0, stream>>>(h,  Wg0, nullptr, xw0,  NN, OUTD, HIDD);
    wmma_gemm<<<dim3(cdiv(NN, 128), OUTD / 64), TB, 0, stream>>>(h,  Wg1, nullptr, xw1,  NN, OUTD, HIDD);
    wmma_gemm<<<dim3(cdiv(NN, 128), OUTD / 64), TB, 0, stream>>>(h1, Wai, nullptr, t,    NN, OUTD, GATF);
    wmma_gemm<<<dim3(cdiv(NN, 128), OUTD / 64), TB, 0, stream>>>(h1, War, nullptr, root, NN, OUTD, GATF);
    // 9) degrees
    deg_kernel<<<cdiv(EE, TB), TB, 0, stream>>>(dst, ew, deg, dega);
    // 10) normalized scatters
    gcn_accum_kernel<<<cdiv(EE + NN, 8), TB, 0, stream>>>(src, dst, ew, deg, xw0, xw1, agg0, agg1);
    arma_accum_kernel<<<cdiv(EE, 8), TB, 0, stream>>>(src, dst, ew, dega, t, agga);
    // 11) concat + activations + tanh
    final_kernel<<<cdiv((size_t)NN * CATF, TB), TB, 0, stream>>>(
        h1, agg0, agg1, agga, root, bg0, bg1, ba, out);
}
